// WaveNetEncoder_79147657331023
// MI455X (gfx1250) — matrix operand order repaired and checked by reading, not compiled, not run on hardware
//
#include <hip/hip_runtime.h>
#include <hip/hip_bf16.h>

// ---------------- problem constants ----------------
#define BSZ   8
#define LEN   90000
#define CH    32
#define NB    6
#define TILE  192                 // output timesteps per workgroup
#define HALO  64                  // left halo: sum(dilations)=63 (+1 spare)
#define TOT   (TILE + HALO)       // 256 timesteps held in LDS
#define DOWNL 22500
#define FLATN (64 * DOWNL)        // 1,440,000
#define NT    ((LEN + TILE - 1) / TILE)   // 469

typedef _Float16 v8h  __attribute__((ext_vector_type(8)));
typedef _Float16 v16h __attribute__((ext_vector_type(16)));
typedef float    v8f  __attribute__((ext_vector_type(8)));
typedef float    v2f  __attribute__((ext_vector_type(2)));
typedef unsigned int u32x4 __attribute__((ext_vector_type(4)));
typedef int          i32x4 __attribute__((ext_vector_type(4)));
typedef int          i32x8 __attribute__((ext_vector_type(8)));

#define WMMA_F16(A, B, Cacc) \
  __builtin_amdgcn_wmma_f32_16x16x32_f16(false, (A), false, (B), (short)0, (Cacc), false, false)

// branch-free fast activations (v_exp_f32 + v_rcp_f32; correct saturation)
__device__ __forceinline__ float fast_tanh(float x) {
  return 1.0f - 2.0f * __builtin_amdgcn_rcpf(__expf(2.0f * x) + 1.0f);
}
__device__ __forceinline__ float fast_sigmoid(float x) {
  return __builtin_amdgcn_rcpf(1.0f + __expf(-x));
}

// A fragment (16x32 f16, M=time, K=chan) per ISA 7.12.2:
// lane<16: K {0..7, 16..23}; lane>=16: K {8..15, 24..31}; row = per-lane.
__device__ __forceinline__ v16h frag_a_row(const _Float16* row, int lane) {
  int kb = (lane >> 4) << 3;                    // 0 or 8
  v8h lo = *(const v8h*)(row + kb);
  v8h hi = *(const v8h*)(row + kb + 16);
  v16h a;
#pragma unroll
  for (int i = 0; i < 8; ++i) { a[i] = lo[i]; a[8 + i] = hi[i]; }
  return a;
}

// B fragment (32x16 f16, K=cin, N=cout), W stored [cout][cin] (cin contig):
// lane holds column n=lane&15 over K {0..7,16..23} (lane<16) / {8..15,24..31} (lane>=16).
__device__ __forceinline__ v16h frag_b_w(const _Float16* W, int co0, int lane) {
  const _Float16* p = W + (co0 + (lane & 15)) * CH + ((lane >> 4) << 3);
  v8h lo = *(const v8h*)(p);
  v8h hi = *(const v8h*)(p + 16);
  v16h b;
#pragma unroll
  for (int i = 0; i < 8; ++i) { b[i] = lo[i]; b[8 + i] = hi[i]; }
  return b;
}

// ---------------- fused WaveNet kernel ----------------
// One workgroup = one (batch, 192-timestep tile). h and skip live in LDS for
// all 6 blocks; halo region is recomputed per block (garbage stays < local 63,
// proven never to reach the output region). y (downsampled skip) goes to ws.
__global__ __launch_bounds__(256) void wavenet_fused(
    const float* __restrict__ x,  const float* __restrict__ w0, const float* __restrict__ b0,
    const float* __restrict__ Wf, const float* __restrict__ bf,
    const float* __restrict__ Wg, const float* __restrict__ bg,
    const float* __restrict__ Ws, const float* __restrict__ bs,
    const float* __restrict__ Wr, const float* __restrict__ br,
    const float* __restrict__ wd, const float* __restrict__ bd,
    float* __restrict__ y)
{
  __shared__ alignas(16) _Float16 hS[TOT][CH];        // 16 KB
  __shared__ alignas(16) float    skipS[TILE][CH];    // 24 KB
  __shared__ alignas(16) _Float16 outS[8][16][CH];    //  8 KB (per-wave staging)
  __shared__ alignas(16) _Float16 WfS[2][CH][CH];     //  4 KB
  __shared__ alignas(16) _Float16 WgS[2][CH][CH];     //  4 KB
  __shared__ alignas(16) _Float16 WsS[CH][CH];        //  2 KB
  __shared__ alignas(16) _Float16 WrS[CH][CH];        //  2 KB
  __shared__ alignas(16) float    xS[TOT + 2];

  const int tid  = threadIdx.x;
  const int lane = tid & 31;
  const int wv   = tid >> 5;
  const int b    = blockIdx.y;
  const long t0  = (long)blockIdx.x * TILE;
  const long gst = t0 - HALO - 2;                 // first x sample needed

  // ---- stage x tile ----
  if (gst >= 0) {
    // interior tile: 1-D TDM load, right-edge OOB zero-fill via tensor_dim0.
    if (wv == 0) {
      unsigned long long ga =
          (unsigned long long)(const void*)(x + (long)b * LEN + gst);
      unsigned int ldsoff = (unsigned int)(unsigned long long)(void*)&xS[0];
      unsigned int tdim0  = (unsigned int)(LEN - gst);      // valid elements
      const unsigned int nelem = TOT + 2;

      u32x4 g0;
      g0[0] = 1u;                                           // count=1 (valid D#)
      g0[1] = ldsoff;                                       // lds_addr
      g0[2] = (unsigned int)(ga & 0xffffffffull);           // global_addr[31:0]
      g0[3] = (unsigned int)((ga >> 32) & 0x1ffffffull)     // global_addr[56:32]
              | (2u << 30);                                 // type=2 ("image")
      i32x8 g1;
      g1[0] = (int)(2u << 16);                              // data_size=4B, mask=0
      g1[1] = (int)((tdim0 & 0xffffu) << 16);               // tensor_dim0[15:0]
      g1[2] = (int)((tdim0 >> 16) & 0xffffu);               // tensor_dim0[31:16]
      g1[3] = (int)(nelem << 16);                           // tile_dim0
      g1[4] = 0;                                            // tile_dim1/2 unused
      g1[5] = (int)tdim0;                                   // tensor_dim0_stride lo
      g1[6] = 0;
      g1[7] = 0;
      i32x4 z4 = {0, 0, 0, 0};
#if __clang_major__ >= 23
      i32x8 z8 = {0, 0, 0, 0, 0, 0, 0, 0};
      __builtin_amdgcn_tensor_load_to_lds(g0, g1, z4, z4, z8, 0);
#else
      __builtin_amdgcn_tensor_load_to_lds(g0, g1, z4, z4, 0);
#endif
      __builtin_amdgcn_s_wait_tensorcnt(0);
    }
  } else {
    // first tile: manual load with left zero-padding
    for (int u = tid; u < TOT + 2; u += 256) {
      long g = gst + u;
      xS[u] = (g >= 0 && g < LEN) ? x[(long)b * LEN + g] : 0.0f;
    }
  }
  __syncthreads();

  // input causal conv (1->32, k=3): h[g] from x[g-2..g]; h==0 for g<0
  for (int e = tid; e < TOT * CH; e += 256) {
    int u = e >> 5, c = e & 31;
    long g = t0 - HALO + u;
    float v = 0.0f;
    if (g >= 0)
      v = w0[c*3+0]*xS[u] + w0[c*3+1]*xS[u+1] + w0[c*3+2]*xS[u+2] + b0[c];
    hS[u][c] = (_Float16)v;
  }
  for (int e = tid; e < TILE * CH; e += 256) (&skipS[0][0])[e] = 0.0f;
  __syncthreads();

  for (int blk = 0; blk < NB; ++blk) {
    const int d = 1 << blk;
    // stage f16 weights for this block: [tap][cout][cin] / [cout][cin]
    const float* Wfp = Wf + (long)blk * CH * CH * 2;
    const float* Wgp = Wg + (long)blk * CH * CH * 2;
    const float* Wsp = Ws + (long)blk * CH * CH;
    const float* Wrp = Wr + (long)blk * CH * CH;
    for (int e = tid; e < CH * CH * 2; e += 256) {
      int co = e >> 6, ci = (e >> 1) & 31, tap = e & 1;
      WfS[tap][co][ci] = (_Float16)Wfp[e];
      WgS[tap][co][ci] = (_Float16)Wgp[e];
    }
    for (int e = tid; e < CH * CH; e += 256) {
      int co = e >> 5, ci = e & 31;
      WsS[co][ci] = (_Float16)Wsp[e];
      WrS[co][ci] = (_Float16)Wrp[e];
    }
    __syncthreads();

    const int m = lane & 15;
    float rsave[2][16];                    // h-increment kept in regs until barrier

#pragma unroll
    for (int tp = 0; tp < 2; ++tp) {       // wave handles 16-time tiles wv and wv+8
      const int tb = (wv + 8 * tp) * 16;
      int u  = tb + m;
      int us = u - d; if (us < 0) us = 0;  // halo garbage never reaches outputs
      v16h a0 = frag_a_row(&hS[us][0], lane);   // tap0: h[t-d]
      v16h a1 = frag_a_row(&hS[u][0],  lane);   // tap1: h[t]

#pragma unroll
      for (int nh = 0; nh < 2; ++nh) {
        const int co0 = nh * 16;
        const int co  = co0 + m;
        v8f facc = {}; v8f gacc = {};
        facc = WMMA_F16(a0, frag_b_w(&WfS[0][0][0], co0, lane), facc);
        facc = WMMA_F16(a1, frag_b_w(&WfS[1][0][0], co0, lane), facc);
        gacc = WMMA_F16(a0, frag_b_w(&WgS[0][0][0], co0, lane), gacc);
        gacc = WMMA_F16(a1, frag_b_w(&WgS[1][0][0], co0, lane), gacc);
        const float bfv = bf[blk*CH + co];
        const float bgv = bg[blk*CH + co];
#pragma unroll
        for (int r = 0; r < 8; ++r) {
          int mm = r + ((lane >> 4) << 3);
          float fv = fast_tanh(facc[r] + bfv);
          float gv = fast_sigmoid(gacc[r] + bgv);
          outS[wv][mm][co] = (_Float16)(fv * gv);   // gated activation, f16 staging
        }
      }
      asm volatile("s_wait_dscnt 0" ::: "memory");  // in-wave LDS RAW (out stage)

      v16h ao = frag_a_row(&outS[wv][m][0], lane);
#pragma unroll
      for (int nh = 0; nh < 2; ++nh) {
        const int co0 = nh * 16;
        const int co  = co0 + m;
        v8f sacc = {}; v8f racc = {};
        sacc = WMMA_F16(ao, frag_b_w(&WsS[0][0], co0, lane), sacc);
        racc = WMMA_F16(ao, frag_b_w(&WrS[0][0], co0, lane), racc);
        const float bsv = bs[blk*CH + co];
        const float brv = br[blk*CH + co];
        if (tb >= HALO) {                 // skip only accumulates real outputs
#pragma unroll
          for (int r = 0; r < 8; ++r) {
            int mm = r + ((lane >> 4) << 3);
            skipS[tb - HALO + mm][co] += sacc[r] + bsv;
          }
        }
#pragma unroll
        for (int r = 0; r < 8; ++r) rsave[tp][nh * 8 + r] = racc[r] + brv;
      }
    }
    __syncthreads();                       // all hS reads done before update

#pragma unroll
    for (int tp = 0; tp < 2; ++tp) {       // h = h + r (keep h==0 for g<0)
      const int tb = (wv + 8 * tp) * 16;
#pragma unroll
      for (int nh = 0; nh < 2; ++nh) {
        const int co = nh * 16 + m;
#pragma unroll
        for (int r = 0; r < 8; ++r) {
          int mm = r + ((lane >> 4) << 3);
          int u  = tb + mm;
          long g = t0 - HALO + u;
          if (g >= 0)
            hS[u][co] = (_Float16)((float)hS[u][co] + rsave[tp][nh * 8 + r]);
        }
      }
    }
    __syncthreads();                       // before next block's reads / W reload
  }

  // downsample conv (stride 4 == kernel 4, non-overlapping) + store y
  for (int e = tid; e < 64 * (TILE / 4); e += 256) {
    int oc = e / (TILE / 4);
    int jj = e % (TILE / 4);
    long col = t0 / 4 + jj;
    if (col < DOWNL) {
      float acc = bd[oc];
#pragma unroll
      for (int q = 0; q < 4; ++q) {
        int u = 4 * jj + q;
        const float* wdp = wd + ((long)oc * CH) * 4 + q;
        for (int c = 0; c < CH; ++c)
          acc += skipS[u][c] * wdp[c * 4];
      }
      y[((long)b * 64 + oc) * DOWNL + col] = acc;
    }
  }
}

// ---------------- final FC: o[8,64] = y_flat @ fcW^T (f32 WMMA) ----------------
// M=16 (8 real batch rows), N=16 per wave (4 oc-tiles), K=4 per instruction.
// fcW (368 MB) streamed exactly once; partial tiles atomically reduced.
__global__ __launch_bounds__(256) void fc_wmma(
    const float* __restrict__ y, const float* __restrict__ fcW, float* __restrict__ oacc)
{
  const int lane = threadIdx.x & 31;
  const int wv   = threadIdx.x >> 5;
  const int co0  = (wv & 3) * 16;        // oc tile
  const int hi   = lane >> 4;
  const int n    = lane & 15;            // = M row (batch) for A, N col (oc) for B
  long kstart = (long)blockIdx.x * 8192 + (long)(wv >> 2) * 4096;
  long kend   = kstart + 4096;
  long rowb   = (long)(co0 + n) * FLATN;

  v8f acc = {};
  for (long k0 = kstart; k0 < kend; k0 += 4) {
    if ((k0 & 1023) == 0)
      __builtin_prefetch(fcW + rowb + k0 + 4096, 0, 1);   // global_prefetch_b8
    // A (16x4 f32): v_j = y[batch][k0 + 2*hi + j]; rows >=8 are zero padding
    v2f a; a[0] = 0.0f; a[1] = 0.0f;
    long ka = k0 + 2 * hi;
    if (n < 8) {
      long base = (long)n * FLATN + ka;
      if (ka     < FLATN) a[0] = y[base];
      if (ka + 1 < FLATN) a[1] = y[base + 1];
    }
    // B (4x16 f32): v_j holds K = k0 + 2*j + hi, column co0+n
    v2f bb; bb[0] = 0.0f; bb[1] = 0.0f;
    long kb0 = k0 + hi, kb1 = k0 + 2 + hi;
    if (kb0 < FLATN) bb[0] = fcW[rowb + kb0];
    if (kb1 < FLATN) bb[1] = fcW[rowb + kb1];
    acc = __builtin_amdgcn_wmma_f32_16x16x4_f32(false, a, false, bb, (short)0, acc, false, false);
  }
  // D[m][n]: m = r + 8*hi (batch), n = lane&15 (oc). Only lanes<16 carry b<8.
  if (lane < 16) {
    int oc = co0 + n;
#pragma unroll
    for (int r = 0; r < 8; ++r)
      atomicAdd(&oacc[r * 64 + oc], acc[r]);
  }
}

__global__ void zero_acc(float* __restrict__ oacc) {
  int i = threadIdx.x;
  if (i < 512) oacc[i] = 0.0f;
}

// mu/logvar split + reparameterization; outputs (mu, logvar, z) concatenated
__global__ void finalize_k(const float* __restrict__ oacc, const float* __restrict__ fcb,
                           const float* __restrict__ eps, float* __restrict__ out)
{
  int i = threadIdx.x;
  if (i < 256) {
    int b = i >> 5, c = i & 31;
    float mu = oacc[b * 64 + c]      + fcb[c];
    float lv = oacc[b * 64 + 32 + c] + fcb[32 + c];
    float z  = mu + eps[b * 32 + c] * __expf(0.5f * lv);
    out[i]       = mu;
    out[256 + i] = lv;
    out[512 + i] = z;
  }
}

extern "C" void kernel_launch(void* const* d_in, const int* in_sizes, int n_in,
                              void* d_out, int out_size, void* d_ws, size_t ws_size,
                              hipStream_t stream) {
  (void)in_sizes; (void)n_in; (void)out_size; (void)ws_size;
  const float* x   = (const float*)d_in[0];
  const float* eps = (const float*)d_in[1];
  const float* w0  = (const float*)d_in[2];
  const float* b0  = (const float*)d_in[3];
  const float* Wf  = (const float*)d_in[4];
  const float* bf  = (const float*)d_in[5];
  const float* Wg  = (const float*)d_in[6];
  const float* bg  = (const float*)d_in[7];
  const float* Ws  = (const float*)d_in[8];
  const float* bs  = (const float*)d_in[9];
  const float* Wr  = (const float*)d_in[10];
  const float* br  = (const float*)d_in[11];
  const float* wd  = (const float*)d_in[12];
  const float* bd  = (const float*)d_in[13];
  const float* fcW = (const float*)d_in[14];
  const float* fcb = (const float*)d_in[15];

  float* oacc = (float*)d_ws;             // 512 floats (padded to 1024)
  float* yws  = (float*)d_ws + 1024;      // 8*64*22500 f32 = 46.08 MB

  zero_acc<<<1, 512, 0, stream>>>(oacc);

  dim3 grid(NT, BSZ);
  wavenet_fused<<<grid, 256, 0, stream>>>(x, w0, b0, Wf, bf, Wg, bg,
                                          Ws, bs, Wr, br, wd, bd, yws);

  int fcblocks = (FLATN + 8191) / 8192;   // 176
  fc_wmma<<<fcblocks, 256, 0, stream>>>(yws, fcW, oacc);

  finalize_k<<<1, 256, 0, stream>>>(oacc, fcb, eps, (float*)d_out);
}
